// L3TransformerConv_84859963834421
// MI455X (gfx1250) — matrix operand order repaired and checked by reading, not compiled, not run on hardware
//
#include <hip/hip_runtime.h>
#include <math.h>

#define N_NODES 20000
#define N_EDGES 320000

typedef __bf16 v16bf __attribute__((ext_vector_type(16)));
typedef __bf16 v8bf  __attribute__((ext_vector_type(8)));
typedef float  v8f   __attribute__((ext_vector_type(8)));

// ---------- helpers ----------

__device__ __forceinline__ __bf16 f2bf(float x) {
    unsigned u = __float_as_uint(x);
    u += 0x7FFFu + ((u >> 16) & 1u);          // round-to-nearest-even
    unsigned short h = (unsigned short)(u >> 16);
    return __builtin_bit_cast(__bf16, h);
}

// float atomic max via int/uint ordering trick (monotone under IEEE-754 bits)
__device__ __forceinline__ void atomicMaxFloat(float* addr, float v) {
    if (!(v < 0.0f)) {
        atomicMax(reinterpret_cast<int*>(addr), __float_as_int(v));
    } else {
        atomicMin(reinterpret_cast<unsigned int*>(addr), __float_as_uint(v));
    }
}

// ---------- cast / pack kernels ----------

// H[M x Kreal] (fp32, packed) -> Hb[M x Kpad] (bf16, zero padded K)
__global__ void cast_act_bf16_kernel(const float* __restrict__ H,
                                     __bf16* __restrict__ Hb,
                                     int M, int Kreal, int Kpad) {
    int i = blockIdx.x * blockDim.x + threadIdx.x;
    int total = M * Kpad;
    if (i >= total) return;
    int row = i / Kpad;
    int k   = i - row * Kpad;
    float x = (k < Kreal) ? H[(size_t)row * Kreal + k] : 0.0f;
    Hb[i] = f2bf(x);
}

// W[di x dout] (fp32, row major) -> Wt[Nrows x Kpad] (bf16) with Wt[n][k] = W[k][n]
__global__ void cast_transpose_w_kernel(const float* __restrict__ W,
                                        __bf16* __restrict__ Wt,
                                        int di, int dout, int Kpad, int Nrows) {
    int i = blockIdx.x * blockDim.x + threadIdx.x;
    int total = Nrows * Kpad;
    if (i >= total) return;
    int n = i / Kpad;
    int k = i - n * Kpad;
    float x = (n < dout && k < di) ? W[(size_t)k * dout + n] : 0.0f;
    Wt[i] = f2bf(x);
}

// ---------- WMMA GEMM: C[M x ncols] = A[M x Kpad] * Wt^T + bias ----------
// One wave computes one 16x16 tile of C. 4 waves per 128-thread block.
__global__ void __launch_bounds__(128)
wmma_gemm_bias_kernel(const __bf16* __restrict__ A,   // [M x Kpad] bf16 (row major)
                      const __bf16* __restrict__ Bt,  // [Nrows x Kpad] bf16 (= B transposed)
                      const float*  __restrict__ bias,// [ncols]
                      float*        __restrict__ C,   // [M x ldc]
                      int M, int Kpad, int ncols, int ldc, int ntilesM) {
    int wave = threadIdx.x >> 5;
    int lane = threadIdx.x & 31;
    int tm = blockIdx.x * 4 + wave;     // uniform per wave -> EXEC stays all-1s
    int tn = blockIdx.y;
    if (tm >= ntilesM) return;

    int g = lane >> 4;                  // lane group (0 or 1)
    int r = lane & 15;

    const __bf16* Arow = A  + (size_t)(tm * 16 + r) * Kpad;
    const __bf16* Brow = Bt + (size_t)(tn * 16 + r) * Kpad;

    v8f acc = {};
    for (int k0 = 0; k0 < Kpad; k0 += 32) {
        // A operand: lane holds row M=r of A; K = {0..7,16..23} + 8*g
        v8bf a0 = *reinterpret_cast<const v8bf*>(Arow + k0 + 8 * g);
        v8bf a1 = *reinterpret_cast<const v8bf*>(Arow + k0 + 16 + 8 * g);
        // B operand: lane holds column N=r of B (row r of Bt); K = {0..7,16..23} + 8*g
        v8bf b0 = *reinterpret_cast<const v8bf*>(Brow + k0 + 8 * g);
        v8bf b1 = *reinterpret_cast<const v8bf*>(Brow + k0 + 16 + 8 * g);

        v16bf a, b;
#pragma unroll
        for (int i = 0; i < 8; ++i) {
            a[i] = a0[i]; a[i + 8] = a1[i];
            b[i] = b0[i]; b[i + 8] = b1[i];
        }
        acc = __builtin_amdgcn_wmma_f32_16x16x32_bf16(
            /*neg_a=*/false, a, /*neg_b=*/false, b,
            /*c_mod=*/(short)0, acc, /*reuse_a=*/false, /*reuse_b=*/false);
    }

    // D layout: VGPR v -> row (v + 8*g), lane r -> column
    int col = tn * 16 + r;
    if (col < ncols) {
        float bv = bias[col];
#pragma unroll
        for (int v = 0; v < 8; ++v) {
            int row = tm * 16 + v + 8 * g;
            if (row < M) C[(size_t)row * ldc + col] = acc[v] + bv;
        }
    }
}

// ---------- edge-phase kernels ----------

__global__ void init_ms_kernel(float* __restrict__ m, float* __restrict__ s, int n) {
    int i = blockIdx.x * blockDim.x + threadIdx.x;
    if (i >= n) return;
    m[i] = -__builtin_inff();
    s[i] = 0.0f;
}

// one wave per edge: logits[e] = scale * dot(Q[dst], K[src]); atomic segment max
__global__ void edge_logits_kernel(const float* __restrict__ Q,
                                   const float* __restrict__ K,
                                   const long long* __restrict__ ei,
                                   float* __restrict__ logits,
                                   float* __restrict__ m,
                                   int E, int d, int ld, float scale) {
    int w = blockIdx.x * (blockDim.x >> 5) + (threadIdx.x >> 5);
    int lane = threadIdx.x & 31;
    if (w >= E) return;
    int src = (int)ei[w];
    int dst = (int)ei[E + w];
    const float* q = Q + (size_t)dst * ld;
    const float* k = K + (size_t)src * ld;
    float acc = 0.0f;
    for (int f = lane; f < d; f += 32) acc += q[f] * k[f];
#pragma unroll
    for (int off = 16; off > 0; off >>= 1) acc += __shfl_xor(acc, off, 32);
    acc *= scale;
    if (lane == 0) {
        logits[w] = acc;
        atomicMaxFloat(&m[dst], acc);
    }
}

// one thread per edge: e = exp(logit - m[dst]); segment sum into s
__global__ void edge_exp_sum_kernel(float* __restrict__ logits,
                                    const long long* __restrict__ ei,
                                    const float* __restrict__ m,
                                    float* __restrict__ s, int E) {
    int e = blockIdx.x * blockDim.x + threadIdx.x;
    if (e >= E) return;
    int dst = (int)ei[E + e];
    float x = __expf(logits[e] - m[dst]);
    logits[e] = x;
    atomicAdd(&s[dst], x);
}

// one wave per edge: Out[dst] += (e/s[dst]) * V[src]
__global__ void edge_agg_kernel(const float* __restrict__ V,
                                const float* __restrict__ expv,
                                const long long* __restrict__ ei,
                                const float* __restrict__ s,
                                float* __restrict__ Out,
                                int E, int d, int ldv, int ldo) {
    int w = blockIdx.x * (blockDim.x >> 5) + (threadIdx.x >> 5);
    int lane = threadIdx.x & 31;
    if (w >= E) return;
    int src = (int)ei[w];
    int dst = (int)ei[E + w];
    float alpha = expv[w] / s[dst];
    const float* v = V + (size_t)src * ldv;
    float* o = Out + (size_t)dst * ldo;
    for (int f = lane; f < d; f += 32) atomicAdd(&o[f], alpha * v[f]);
}

__global__ void relu_kernel(float* __restrict__ x, int n) {
    int i = blockIdx.x * blockDim.x + threadIdx.x;
    if (i >= n) return;
    x[i] = fmaxf(x[i], 0.0f);
}

// ---------- host-side layer driver ----------

static void run_layer(const float* Hin, int di, int dout,
                      const float* Wq, const float* bq,
                      const float* Wk, const float* bk,
                      const float* Wv, const float* bv,
                      const float* Ws, const float* bs,
                      const long long* ei,
                      float* Out,
                      __bf16* Hb, __bf16* WtQ, __bf16* WtK, __bf16* WtV, __bf16* WtS,
                      float* Q, float* Kf, float* V,
                      float* logits, float* m, float* s,
                      hipStream_t stream) {
    const int Kpad  = (di + 31) & ~31;
    const int ntN   = (dout + 15) / 16;
    const int Nrows = ntN * 16;
    const int ntM   = N_NODES / 16;   // 20000 / 16 = 1250 exactly

    {   // activations -> bf16 padded
        int total = N_NODES * Kpad;
        cast_act_bf16_kernel<<<(total + 255) / 256, 256, 0, stream>>>(Hin, Hb, N_NODES, di, Kpad);
    }
    {   // weights -> bf16, transposed + padded
        int total = Nrows * Kpad;
        int blocks = (total + 255) / 256;
        cast_transpose_w_kernel<<<blocks, 256, 0, stream>>>(Wq, WtQ, di, dout, Kpad, Nrows);
        cast_transpose_w_kernel<<<blocks, 256, 0, stream>>>(Wk, WtK, di, dout, Kpad, Nrows);
        cast_transpose_w_kernel<<<blocks, 256, 0, stream>>>(Wv, WtV, di, dout, Kpad, Nrows);
        cast_transpose_w_kernel<<<blocks, 256, 0, stream>>>(Ws, WtS, di, dout, Kpad, Nrows);
    }
    dim3 gg((ntM + 3) / 4, ntN);
    wmma_gemm_bias_kernel<<<gg, 128, 0, stream>>>(Hb, WtQ, bq, Q,   N_NODES, Kpad, dout, dout, ntM);
    wmma_gemm_bias_kernel<<<gg, 128, 0, stream>>>(Hb, WtK, bk, Kf,  N_NODES, Kpad, dout, dout, ntM);
    wmma_gemm_bias_kernel<<<gg, 128, 0, stream>>>(Hb, WtV, bv, V,   N_NODES, Kpad, dout, dout, ntM);
    wmma_gemm_bias_kernel<<<gg, 128, 0, stream>>>(Hb, WtS, bs, Out, N_NODES, Kpad, dout, dout, ntM); // skip branch

    init_ms_kernel<<<(N_NODES + 255) / 256, 256, 0, stream>>>(m, s, N_NODES);

    float scale = 1.0f / sqrtf((float)dout);
    edge_logits_kernel<<<(N_EDGES + 7) / 8, 256, 0, stream>>>(Q, Kf, ei, logits, m, N_EDGES, dout, dout, scale);
    edge_exp_sum_kernel<<<(N_EDGES + 255) / 256, 256, 0, stream>>>(logits, ei, m, s, N_EDGES);
    edge_agg_kernel<<<(N_EDGES + 7) / 8, 256, 0, stream>>>(V, logits, ei, s, Out, N_EDGES, dout, dout, dout);

    relu_kernel<<<((N_NODES * dout) + 255) / 256, 256, 0, stream>>>(Out, N_NODES * dout);
}

// ---------- entry ----------

extern "C" void kernel_launch(void* const* d_in, const int* in_sizes, int n_in,
                              void* d_out, int out_size, void* d_ws, size_t ws_size,
                              hipStream_t stream) {
    (void)in_sizes; (void)n_in; (void)out_size; (void)ws_size;

    const float*     x  = (const float*)d_in[0];
    const long long* ei = (const long long*)d_in[1];   // int64 [2, E]

    // dict order: W{q,k,v,s}{1,2,3} each followed by its bias
    const float* W[12]; const float* B[12];
    for (int i = 0; i < 12; ++i) {
        W[i] = (const float*)d_in[2 + 2 * i];
        B[i] = (const float*)d_in[3 + 2 * i];
    }

    // scratch carve (256B aligned slices)
    char* p = (char*)d_ws;
    auto alloc = [&](size_t bytes) -> void* {
        void* r = (void*)p;
        p += (bytes + 255) & ~(size_t)255;
        return r;
    };
    __bf16* Hb  = (__bf16*)alloc((size_t)N_NODES * 416 * sizeof(__bf16));
    __bf16* WtQ = (__bf16*)alloc((size_t)416 * 416 * sizeof(__bf16));
    __bf16* WtK = (__bf16*)alloc((size_t)416 * 416 * sizeof(__bf16));
    __bf16* WtV = (__bf16*)alloc((size_t)416 * 416 * sizeof(__bf16));
    __bf16* WtS = (__bf16*)alloc((size_t)416 * 416 * sizeof(__bf16));
    float*  Q   = (float*)alloc((size_t)N_NODES * 400 * sizeof(float));
    float*  Kf  = (float*)alloc((size_t)N_NODES * 400 * sizeof(float));
    float*  V   = (float*)alloc((size_t)N_NODES * 400 * sizeof(float));
    float*  H1  = (float*)alloc((size_t)N_NODES * 400 * sizeof(float));
    float*  H2  = (float*)alloc((size_t)N_NODES * 200 * sizeof(float));
    float*  lg  = (float*)alloc((size_t)N_EDGES * sizeof(float));
    float*  m   = (float*)alloc((size_t)N_NODES * sizeof(float));
    float*  s   = (float*)alloc((size_t)N_NODES * sizeof(float));

    // layer 1: 128 -> 400
    run_layer(x, 128, 400,
              W[0], B[0], W[1], B[1], W[2], B[2], W[3], B[3],
              ei, H1, Hb, WtQ, WtK, WtV, WtS, Q, Kf, V, lg, m, s, stream);
    // layer 2: 400 -> 200
    run_layer(H1, 400, 200,
              W[4], B[4], W[5], B[5], W[6], B[6], W[7], B[7],
              ei, H2, Hb, WtQ, WtK, WtV, WtS, Q, Kf, V, lg, m, s, stream);
    // layer 3: 200 -> 4, straight into d_out (packed ld = 4)
    run_layer(H2, 200, 4,
              W[8], B[8], W[9], B[9], W[10], B[10], W[11], B[11],
              ei, (float*)d_out, Hb, WtQ, WtK, WtV, WtS, Q, Kf, V, lg, m, s, stream);
}
